// QuantizedLinear_15985868276124
// MI455X (gfx1250) — compile-verified, matrix operand order checked by reading, not run on hardware
//
#include <hip/hip_runtime.h>
#include <hip/hip_bf16.h>

// ---------------------------------------------------------------------------
// QuantizedLinear (AQLM-style, 2 codebooks, group ig=8, og=1) for gfx1250.
//   Pass 1: dequantize W -> bf16 [OUT_F][IN_F] in workspace
//   Pass 2: bf16 WMMA GEMM  out[8192,11008] = x @ W^T + bias (f32 accum)
//           128x256 block, 64x64 wave tiles, double-buffered LDS,
//           W tile staged via GLOBAL_LOAD_ASYNC_TO_LDS_B128 (ASYNCcnt).
// ---------------------------------------------------------------------------

#define IN_F   4096
#define OUT_F  11008
#define TOKENS 8192      // 4 * 2048
#define NI     512       // IN_F / 8
#define CB_SZ  256

#define BM 128
#define BN 256
#define BK 32
#define NK (IN_F / BK)   // 128 K-steps
#define LDSS 40          // LDS row stride in bf16 elements (80B, bank-conflict free)

typedef __attribute__((ext_vector_type(16))) __bf16 v16bf;
typedef __attribute__((ext_vector_type(8)))  float  v8f;

union FragU {
    uint4 u[2];
    v16bf v;
};

// f32 -> bf16 pair (round-half-up), packed into one dword:
// 2x v_add_nc_u32 (dual-issuable) + 1x v_perm_b32
__device__ __forceinline__ unsigned int pk_bf16(float a, float b) {
    unsigned int ua = __builtin_bit_cast(unsigned int, a) + 0x8000u;
    unsigned int ub = __builtin_bit_cast(unsigned int, b) + 0x8000u;
    // bytes: [ub.3, ub.2, ua.3, ua.2] -> low16 = bf16(a), high16 = bf16(b)
    return __builtin_amdgcn_perm(ub, ua, 0x07060302u);
}

// ---------------------------------------------------------------------------
// Pass 1: dequantize. One thread per (no, ni) group of 8 weights.
// ---------------------------------------------------------------------------
__global__ __launch_bounds__(256) void aqlm_dequant_bf16(
    const int*   __restrict__ codes,      // [OUT_F][NI][2]
    const float* __restrict__ codebooks,  // [2][256][1][8]
    const float* __restrict__ scales,     // [OUT_F]
    unsigned short* __restrict__ wq)      // [OUT_F][IN_F] bf16
{
    int gid = blockIdx.x * 256 + threadIdx.x;   // exactly OUT_F*NI threads
    int no  = gid >> 9;                          // / NI
    int c0  = codes[(size_t)gid * 2 + 0];
    int c1  = codes[(size_t)gid * 2 + 1];
    const float* cb0 = codebooks + (size_t)c0 * 8;
    const float* cb1 = codebooks + (size_t)CB_SZ * 8 + (size_t)c1 * 8;
    float s = scales[no];

    unsigned int o[4];
#pragma unroll
    for (int j = 0; j < 4; ++j) {
        float f0 = s * (cb0[2*j]   + cb1[2*j]);
        float f1 = s * (cb0[2*j+1] + cb1[2*j+1]);
        o[j] = pk_bf16(f0, f1);
    }
    uint4 pack; pack.x = o[0]; pack.y = o[1]; pack.z = o[2]; pack.w = o[3];
    *(uint4*)(wq + (size_t)gid * 8) = pack;   // no*IN_F + ni*8 == gid*8
}

// ---------------------------------------------------------------------------
// Pass 2: bf16 WMMA GEMM.
// Block = 256 threads = 8 waves arranged 2(M) x 4(N); wave tile = 64x64
// = 4x4 tiles of 16x16 -> 16 v_wmma_f32_16x16x32_bf16 per K-step.
// ---------------------------------------------------------------------------
__global__ __launch_bounds__(256) void aqlm_wmma_gemm(
    const float*          __restrict__ X,     // [TOKENS][IN_F] f32
    const unsigned short* __restrict__ Wq,    // [OUT_F][IN_F] bf16
    const float*          __restrict__ bias,  // [OUT_F]
    float*                __restrict__ out)   // [TOKENS][OUT_F] f32
{
    __shared__ __align__(16) unsigned short sA[2][BM * LDSS];  // x tile bf16
    __shared__ __align__(16) unsigned short sB[2][BN * LDSS];  // W tile bf16

    const int tid    = threadIdx.x;
    const int lane   = tid & 31;
    const int wid    = tid >> 5;
    const int wave_m = wid & 1;      // 0..1 -> 64-row strip
    const int wave_n = wid >> 1;     // 0..3 -> 64-col strip
    const int lcol   = lane & 15;
    const int lhalf  = lane >> 4;

    const int mBase = blockIdx.y * BM;
    const int nBase = blockIdx.x * BN;

    // ---- B (W) tile async-copy descriptors: 1024 uint4 chunks / 256 thr ----
    // chunk e: row = e>>2 (N within tile), c = e&3 (16B column chunk)
    unsigned b_lds[4];    // LDS byte address (buffer 0)
    unsigned b_off[4];    // byte offset from Wq base (k0 added per step)
#pragma unroll
    for (int i = 0; i < 4; ++i) {
        int e = tid + i * 256;
        int r = e >> 2, c = e & 3;
        b_lds[i] = (unsigned)(size_t)&sB[0][r * LDSS + c * 8];
        b_off[i] = ((unsigned)(nBase + r) * IN_F + (unsigned)c * 8) * 2u;
    }

    // per-wave bias values
    float biasv[4];
#pragma unroll
    for (int nj = 0; nj < 4; ++nj)
        biasv[nj] = bias[nBase + wave_n * 64 + nj * 16 + lcol];

    v8f acc[4][4];
#pragma unroll
    for (int mi = 0; mi < 4; ++mi)
#pragma unroll
        for (int nj = 0; nj < 4; ++nj)
            acc[mi][nj] = (v8f){0.f,0.f,0.f,0.f,0.f,0.f,0.f,0.f};

    // ---------------- helpers ----------------
    auto issueB = [&](int kt, int buf) {
        const unsigned kbyte = (unsigned)kt * BK * 2u;
        const unsigned bufoff = (unsigned)buf * (BN * LDSS * 2u);
#pragma unroll
        for (int i = 0; i < 4; ++i) {
            asm volatile("global_load_async_to_lds_b128 %0, %1, %2"
                         :: "v"(b_lds[i] + bufoff), "v"(b_off[i] + kbyte),
                            "s"(Wq)
                         : "memory");
        }
    };
    auto loadA = [&](int kt, float4* av) {
        const int k0 = kt * BK;
#pragma unroll
        for (int i = 0; i < 4; ++i) {
            int e = tid + i * 256;
            int r = e >> 3, c = e & 7;
            av[i] = *(const float4*)(X + (size_t)(mBase + r) * IN_F + k0 + c * 4);
        }
    };
    auto storeA = [&](const float4* av, int buf) {
#pragma unroll
        for (int i = 0; i < 4; ++i) {
            int e = tid + i * 256;
            int r = e >> 3, c = e & 7;
            uint2 p;
            p.x = pk_bf16(av[i].x, av[i].y);
            p.y = pk_bf16(av[i].z, av[i].w);
            *(uint2*)&sA[buf][r * LDSS + c * 4] = p;
        }
    };

    // ---------------- prologue: stage K-step 0 into buffer 0 ----------------
    issueB(0, 0);
    {
        float4 av[4];
        loadA(0, av);
        storeA(av, 0);
    }

    // ---------------- main loop: one barrier per K-step ----------------
    for (int kt = 0; kt < NK; ++kt) {
        const int cur = kt & 1;
        const int nxt = cur ^ 1;

        // our async copies into `cur` have landed; barrier makes all waves'
        // copies + ds_stores globally visible (syncthreads waits DScnt too)
        asm volatile("s_wait_asynccnt 0x0" ::: "memory");
        __syncthreads();

        const bool pre = (kt + 1) < NK;
        float4 av[4];
        if (pre) {                 // uniform branch: EXEC stays all-ones
            issueB(kt + 1, nxt);   // W tile kt+1 -> LDS (async)
            loadA(kt + 1, av);     // x tile kt+1 -> regs (overlaps compute)
        }

        // ---- fragments from buffer `cur` (ISA 16-bit A/B layout:
        //      lane half h owns K in [8h,8h+8) and [16+8h,16+8h+8)) ----
        FragU fb[4];
#pragma unroll
        for (int nj = 0; nj < 4; ++nj) {
            int col = wave_n * 64 + nj * 16 + lcol;
            fb[nj].u[0] = *(const uint4*)&sB[cur][col * LDSS + lhalf * 8];
            fb[nj].u[1] = *(const uint4*)&sB[cur][col * LDSS + 16 + lhalf * 8];
        }
#pragma unroll
        for (int mi = 0; mi < 4; ++mi) {
            FragU fa;
            int row = wave_m * 64 + mi * 16 + lcol;
            fa.u[0] = *(const uint4*)&sA[cur][row * LDSS + lhalf * 8];
            fa.u[1] = *(const uint4*)&sA[cur][row * LDSS + 16 + lhalf * 8];
#pragma unroll
            for (int nj = 0; nj < 4; ++nj)
                acc[mi][nj] = __builtin_amdgcn_wmma_f32_16x16x32_bf16(
                    false, fa.v, false, fb[nj].v,
                    (short)0, acc[mi][nj], false, false);
        }

        if (pre)
            storeA(av, nxt);       // x tile kt+1 -> LDS buffer `nxt`
    }

    // ---- epilogue: C/D layout: VGPR v, lane l -> M = v + 8*(l>=16), N = l%16
#pragma unroll
    for (int mi = 0; mi < 4; ++mi) {
#pragma unroll
        for (int nj = 0; nj < 4; ++nj) {
            int col = nBase + wave_n * 64 + nj * 16 + lcol;
#pragma unroll
            for (int v = 0; v < 8; ++v) {
                int row = mBase + wave_m * 64 + mi * 16 + lhalf * 8 + v;
                out[(size_t)row * OUT_F + col] = acc[mi][nj][v] + biasv[nj];
            }
        }
    }
}

// ---------------------------------------------------------------------------
extern "C" void kernel_launch(void* const* d_in, const int* in_sizes, int n_in,
                              void* d_out, int out_size, void* d_ws, size_t ws_size,
                              hipStream_t stream) {
    const float* x         = (const float*)d_in[0];
    const int*   codes     = (const int*)  d_in[1];
    const float* codebooks = (const float*)d_in[2];
    const float* scales    = (const float*)d_in[3];
    const float* bias      = (const float*)d_in[4];
    float*       out       = (float*)d_out;
    unsigned short* wq     = (unsigned short*)d_ws;   // 11008*4096*2 = 90.2 MB

    // Pass 1: dequantize weights to bf16
    aqlm_dequant_bf16<<<(OUT_F * NI) / 256, 256, 0, stream>>>(
        codes, codebooks, scales, wq);

    // Pass 2: WMMA GEMM (43 x 64 blocks of 128x256)
    dim3 grid(OUT_F / BN, TOKENS / BM);
    aqlm_wmma_gemm<<<grid, 256, 0, stream>>>(x, wq, bias, out);
}
